// MuGuidedAttention_28346784154027
// MI455X (gfx1250) — matrix operand order repaired and checked by reading, not compiled, not run on hardware
//
#include <hip/hip_runtime.h>

#define TOKS 2048
#define HID  2048
#define NHQ  32
#define NKVH 8
#define HD   64

typedef __attribute__((ext_vector_type(16))) __bf16 v16bf;
typedef __attribute__((ext_vector_type(8)))  float  v8f;
typedef __attribute__((ext_vector_type(4)))  int    v4i_t;

static __device__ __forceinline__ unsigned short f2bf(float f) {
  unsigned int u = __float_as_uint(f);
  unsigned int r = u + 0x7FFFu + ((u >> 16) & 1u);   // round-to-nearest-even
  return (unsigned short)(r >> 16);
}

// ---- CDNA5 async global->LDS copy (ASYNCcnt-tracked), with sync fallback ----
static __device__ __forceinline__ void async_cp16(const void* g, void* l) {
#if __has_builtin(__builtin_amdgcn_global_load_async_to_lds_b128)
  // param 1: v4i in global AS (implicit generic->AS1 cast), param 2: LDS dest
  __builtin_amdgcn_global_load_async_to_lds_b128((v4i_t*)g, (v4i_t*)l, 0, 0);
#else
  *(uint4*)l = *(const uint4*)g;
#endif
}
static __device__ __forceinline__ void wait_async0() {
#if __has_builtin(__builtin_amdgcn_s_wait_asynccnt)
  __builtin_amdgcn_s_wait_asynccnt(0);
#elif __has_builtin(__builtin_amdgcn_global_load_async_to_lds_b128)
  asm volatile("s_wait_asynccnt 0x0" ::: "memory");
#endif
}

union FragU { v16bf v; unsigned short u[16]; };

// A fragment (16x32 bf16): lane l holds row m=l%16; K = kb..kb+15, kb=(l/16)*16.
static __device__ __forceinline__ v16bf load_a_frag(const unsigned short* p0, int stride, int lane) {
  int m  = lane & 15;
  int kb = (lane >> 4) * 16;
  const unsigned short* p = p0 + m * stride + kb;
  FragU f;
#pragma unroll
  for (int e = 0; e < 8; ++e) f.u[e] = p[e];
#pragma unroll
  for (int e = 0; e < 8; ++e) f.u[8 + e] = p[8 + e];
  return f.v;
}

// B fragment (32x16 bf16) from K-contiguous LDS layout Bt[n][k]:
// lane l holds column n=l%16, K=(l/16)*16 + 0..15 (16 contiguous halfs).
static __device__ __forceinline__ v16bf load_b_frag(const unsigned short* p0, int stride, int lane) {
  int n = lane & 15;
  int k = (lane >> 4) * 16;
  const unsigned short* p = p0 + n * stride + k;
  FragU f;
#pragma unroll
  for (int e = 0; e < 16; ++e) f.u[e] = p[e];
  return f.v;
}

// ---------------------------------------------------------------------------
// One-time packing kernels
// ---------------------------------------------------------------------------
__global__ __launch_bounds__(256) void cvt_bf16_kernel(
    const float* __restrict__ src, unsigned short* __restrict__ dst, int n) {
  int i = blockIdx.x * 256 + threadIdx.x;
  if (i < n) dst[i] = f2bf(src[i]);
}

// [R][C] f32 -> [C][R] bf16 (LDS-tiled, coalesced both sides).
__global__ __launch_bounds__(256) void transpose_cvt_kernel(
    const float* __restrict__ src, unsigned short* __restrict__ dst, int R, int C) {
  __shared__ unsigned short t[32][33];
  const int c0 = blockIdx.x * 32, r0 = blockIdx.y * 32;
  const int tc = threadIdx.x & 31, tr8 = threadIdx.x >> 5;  // 8 rows/pass
#pragma unroll
  for (int i = 0; i < 4; ++i) {
    int r = tr8 + i * 8;
    t[r][tc] = f2bf(src[(size_t)(r0 + r) * C + c0 + tc]);
  }
  __syncthreads();
#pragma unroll
  for (int i = 0; i < 4; ++i) {
    int r = tr8 + i * 8;  // output row within tile (= original column)
    dst[(size_t)(c0 + r) * R + r0 + tc] = t[tc][r];
  }
}

// ---------------------------------------------------------------------------
// Generic bf16 GEMM: out[M][N] (f32) = A0[M][2048] @ B0t^T (+ A1 @ B1t^T).
// Bt is pre-packed [N][K=2048] bf16.  Double-buffered async global->LDS
// pipeline: one barrier per K-step, copy of step s+1 overlaps WMMA of step s.
// Block tile 64x64, K_TILE=32, 8 waves (1 M-tile x 2 N-tiles each).
// ---------------------------------------------------------------------------
__global__ __launch_bounds__(256) void gemm_bf16_kernel(
    const unsigned short* __restrict__ A0, const unsigned short* __restrict__ A1,
    const unsigned short* __restrict__ B0t, const unsigned short* __restrict__ B1t,
    float* __restrict__ outp, int N) {
  __shared__ unsigned short As[2][64][32];
  __shared__ unsigned short Bs[2][64][32];
  const int tid = threadIdx.x, lane = tid & 31, wave = tid >> 5;
  const int m0 = blockIdx.x * 64, n0 = blockIdx.y * 64;
  const int mt = wave & 3, nb0 = (wave >> 2) * 2;
  const int nsteps = A1 ? 128 : 64;           // 2 phases x 64 K-steps
  const int am = tid >> 2, ac = (tid & 3) * 8; // one 16B chunk per thread/tile

  v8f acc0, acc1;
#pragma unroll
  for (int i = 0; i < 8; ++i) { acc0[i] = 0.0f; acc1[i] = 0.0f; }

  auto issue = [&](int buf, int step) {
    const unsigned short* A  = (step >> 6) ? A1 : A0;
    const unsigned short* Bt = (step >> 6) ? B1t : B0t;
    int k0 = (step & 63) * 32;
    async_cp16(A + (size_t)(m0 + am) * HID + k0 + ac, &As[buf][am][ac]);
    async_cp16(Bt + (size_t)(n0 + am) * HID + k0 + ac, &Bs[buf][am][ac]);
  };

  issue(0, 0);
  for (int s = 0; s < nsteps; ++s) {
    const int c = s & 1;
    wait_async0();            // current buffer landed (next not yet issued)
    __syncthreads();          // all waves' copies visible; prev reads retired
    if (s + 1 < nsteps) issue(1 - c, s + 1);   // overlaps with WMMA below
    v16bf a  = load_a_frag(&As[c][mt * 16][0], 32, lane);
    v16bf b0 = load_b_frag(&Bs[c][nb0 * 16][0], 32, lane);
    v16bf b1 = load_b_frag(&Bs[c][(nb0 + 1) * 16][0], 32, lane);
    acc0 = __builtin_amdgcn_wmma_f32_16x16x32_bf16(false, a, false, b0, (short)0, acc0, false, false);
    acc1 = __builtin_amdgcn_wmma_f32_16x16x32_bf16(false, a, false, b1, (short)0, acc1, false, false);
  }

  const int hi = lane >> 4, n = lane & 15;
#pragma unroll
  for (int r = 0; r < 8; ++r) {
    int row = m0 + mt * 16 + r + 8 * hi;
    outp[(size_t)row * N + n0 + nb0 * 16 + n]       = acc0[r];
    outp[(size_t)row * N + n0 + (nb0 + 1) * 16 + n] = acc1[r];
  }
}

// ---------------------------------------------------------------------------
// Per-head RMSNorm + RoPE, f32 -> bf16.  One wave per (token, head).
// ---------------------------------------------------------------------------
__global__ __launch_bounds__(32) void norm_rope_kernel(
    const float* __restrict__ src, unsigned short* __restrict__ dst,
    const float* __restrict__ w, const int* __restrict__ positions, int nheads) {
  const int bid = blockIdx.x;
  const int t = bid / nheads, hh = bid % nheads;
  const int j = threadIdx.x;
  const float* row = src + (size_t)t * nheads * HD + hh * HD;
  float x1 = row[j], x2 = row[j + 32];
  float ss = x1 * x1 + x2 * x2;
#pragma unroll
  for (int m = 16; m >= 1; m >>= 1) ss += __shfl_xor(ss, m);
  float rr = rsqrtf(ss * (1.0f / 64.0f) + 1e-6f);
  float n1 = x1 * rr * w[j], n2 = x2 * rr * w[j + 32];
  float pos = (float)positions[t];
  float inv = __expf(-0.28782313662425572f * (float)j);  // 10000^(-2j/64)
  float f = pos * inv;
  float s, c;
  __sincosf(f, &s, &c);
  unsigned short* orow = dst + (size_t)t * nheads * HD + hh * HD;
  orow[j]      = f2bf(n1 * c - n2 * s);
  orow[j + 32] = f2bf(n2 * c + n1 * s);
}

// ---------------------------------------------------------------------------
// Causal flash attention with GQA, bf16 WMMA, async double-buffered K/V tiles.
// vbT is pre-packed [NKVH*HD][TOKS] bf16 so V^T tiles are contiguous copies.
// grid = (T/64 q-blocks, 32 heads), 4 waves; wave w owns q rows q0+16w..+15.
// ---------------------------------------------------------------------------
__global__ __launch_bounds__(128) void flash_attn_kernel(
    const unsigned short* __restrict__ qb, const unsigned short* __restrict__ kb,
    const unsigned short* __restrict__ vbT, unsigned short* __restrict__ ob) {
  __shared__ unsigned short Qs[64][64];        // [q][d]
  __shared__ unsigned short Ks[2][64][64];     // [key][d]  (Bt for Q@K^T)
  __shared__ unsigned short Vt[2][64][64];     // [d][key]  (Bt for P@V)
  __shared__ unsigned short Ps[4][16][64];     // per-wave P strip [q][key]

  const int tid = threadIdx.x, lane = tid & 31, wave = tid >> 5;
  const int h = blockIdx.y, kvh = h >> 2;
  const int q0 = blockIdx.x * 64;
  const int hi = lane >> 4, nn = lane & 15;
  const float scale = 0.125f;                  // 1/sqrt(64)

  auto issue_kv = [&](int buf, int j0) {
#pragma unroll
    for (int i = 0; i < 4; ++i) {              // 8KB per tile, 16B per chunk
      int idx = tid + i * 128;
      int r = idx >> 3, cc = (idx & 7) * 8;
      async_cp16(kb  + (size_t)(j0 + r) * (NKVH * HD) + kvh * HD + cc, &Ks[buf][r][cc]);
      async_cp16(vbT + (size_t)(kvh * HD + r) * TOKS + j0 + cc,        &Vt[buf][r][cc]);
    }
  };

#pragma unroll
  for (int i = 0; i < 4; ++i) {                // Q tile (async, drained by 1st wait)
    int idx = tid + i * 128;
    int r = idx >> 3, cc = (idx & 7) * 8;
    async_cp16(qb + (size_t)(q0 + r) * (NHQ * HD) + h * HD + cc, &Qs[r][cc]);
  }
  issue_kv(0, 0);

  float mrow[8], lrow[8];
  v8f O[4];
#pragma unroll
  for (int r = 0; r < 8; ++r) { mrow[r] = -3.0e38f; lrow[r] = 0.0f; }
#pragma unroll
  for (int nt = 0; nt < 4; ++nt)
#pragma unroll
    for (int r = 0; r < 8; ++r) O[nt][r] = 0.0f;

  for (int j0 = 0; j0 <= q0; j0 += 64) {
    const int c = (j0 >> 6) & 1;
    wait_async0();
    __syncthreads();
    if (j0 + 64 <= q0) issue_kv(1 - c, j0 + 64);  // overlaps compute below

    // S = Q K^T (scaled, causal-masked)
    float s[4][8];
#pragma unroll
    for (int nt = 0; nt < 4; ++nt) {
      v8f sa;
#pragma unroll
      for (int i = 0; i < 8; ++i) sa[i] = 0.0f;
#pragma unroll
      for (int ks = 0; ks < 2; ++ks) {
        v16bf a = load_a_frag(&Qs[wave * 16][ks * 32], 64, lane);
        v16bf b = load_b_frag(&Ks[c][nt * 16][ks * 32], 64, lane);
        sa = __builtin_amdgcn_wmma_f32_16x16x32_bf16(false, a, false, b, (short)0, sa, false, false);
      }
      int kcol = j0 + nt * 16 + nn;
#pragma unroll
      for (int r = 0; r < 8; ++r) {
        int qrow = q0 + wave * 16 + r + 8 * hi;
        s[nt][r] = (kcol > qrow) ? -3.0e38f : sa[r] * scale;
      }
    }

    // online softmax (row stats across the 16 lanes of each half-wave)
#pragma unroll
    for (int r = 0; r < 8; ++r) {
      float mc = s[0][r];
#pragma unroll
      for (int nt = 1; nt < 4; ++nt) mc = fmaxf(mc, s[nt][r]);
#pragma unroll
      for (int m = 8; m >= 1; m >>= 1) mc = fmaxf(mc, __shfl_xor(mc, m));
      float mn = fmaxf(mrow[r], mc);
      float al = __expf(mrow[r] - mn);
      float psum = 0.0f;
#pragma unroll
      for (int nt = 0; nt < 4; ++nt) {
        float p = __expf(s[nt][r] - mn);
        s[nt][r] = p;
        psum += p;
      }
#pragma unroll
      for (int m = 8; m >= 1; m >>= 1) psum += __shfl_xor(psum, m);
      lrow[r] = lrow[r] * al + psum;
      mrow[r] = mn;
#pragma unroll
      for (int nt = 0; nt < 4; ++nt) O[nt][r] *= al;
    }

    // C-layout -> A-layout via per-wave LDS strip (same-wave LDS is in-order)
#pragma unroll
    for (int nt = 0; nt < 4; ++nt)
#pragma unroll
      for (int r = 0; r < 8; ++r)
        Ps[wave][r + 8 * hi][nt * 16 + nn] = f2bf(s[nt][r]);

    // O += P @ V
#pragma unroll
    for (int nt = 0; nt < 4; ++nt)
#pragma unroll
      for (int ks = 0; ks < 2; ++ks) {
        v16bf a = load_a_frag(&Ps[wave][0][ks * 32], 64, lane);
        v16bf b = load_b_frag(&Vt[c][nt * 16][ks * 32], 64, lane);
        O[nt] = __builtin_amdgcn_wmma_f32_16x16x32_bf16(false, a, false, b, (short)0, O[nt], false, false);
      }
  }

#pragma unroll
  for (int nt = 0; nt < 4; ++nt)
#pragma unroll
    for (int r = 0; r < 8; ++r) {
      int row = q0 + wave * 16 + r + 8 * hi;
      float o = O[nt][r] / lrow[r];
      ob[(size_t)row * (NHQ * HD) + h * HD + nt * 16 + nn] = f2bf(o);
    }
}

// ---------------------------------------------------------------------------
extern "C" void kernel_launch(void* const* d_in, const int* in_sizes, int n_in,
                              void* d_out, int out_size, void* d_ws, size_t ws_size,
                              hipStream_t stream) {
  (void)in_sizes; (void)n_in; (void)out_size; (void)ws_size;
  const float* hidden = (const float*)d_in[0];
  const float* mu     = (const float*)d_in[1];
  const float* Wq     = (const float*)d_in[2];
  const float* Wk     = (const float*)d_in[3];
  const float* Wv     = (const float*)d_in[4];
  const float* Wo     = (const float*)d_in[5];
  const float* Wmq    = (const float*)d_in[6];
  const float* Wmk    = (const float*)d_in[7];
  const float* Wmv    = (const float*)d_in[8];
  const float* qnw    = (const float*)d_in[9];
  const float* knw    = (const float*)d_in[10];
  const int*   pos    = (const int*)d_in[11];
  float* out = (float*)d_out;

  char* ws = (char*)d_ws;
  size_t off = 0;
  float* qf = (float*)(ws + off);                    off += (size_t)TOKS * NHQ * HD * 4;
  float* kf = (float*)(ws + off);                    off += (size_t)TOKS * NKVH * HD * 4;
  float* vf = (float*)(ws + off);                    off += (size_t)TOKS * NKVH * HD * 4;
  unsigned short* qb  = (unsigned short*)(ws + off); off += (size_t)TOKS * NHQ * HD * 2;
  unsigned short* kb  = (unsigned short*)(ws + off); off += (size_t)TOKS * NKVH * HD * 2;
  unsigned short* vbT = (unsigned short*)(ws + off); off += (size_t)TOKS * NKVH * HD * 2;
  unsigned short* ob  = (unsigned short*)(ws + off); off += (size_t)TOKS * NHQ * HD * 2;
  unsigned short* hb  = (unsigned short*)(ws + off); off += (size_t)TOKS * HID * 2;
  unsigned short* mb  = (unsigned short*)(ws + off); off += (size_t)TOKS * HID * 2;
  unsigned short* WqT = (unsigned short*)(ws + off); off += (size_t)HID * NHQ * HD * 2;
  unsigned short* WmqT= (unsigned short*)(ws + off); off += (size_t)HID * NHQ * HD * 2;
  unsigned short* WkT = (unsigned short*)(ws + off); off += (size_t)HID * NKVH * HD * 2;
  unsigned short* WmkT= (unsigned short*)(ws + off); off += (size_t)HID * NKVH * HD * 2;
  unsigned short* WvT = (unsigned short*)(ws + off); off += (size_t)HID * NKVH * HD * 2;
  unsigned short* WmvT= (unsigned short*)(ws + off); off += (size_t)HID * NKVH * HD * 2;
  unsigned short* WoT = (unsigned short*)(ws + off); off += (size_t)NHQ * HD * HID * 2;

  // 0) one-time pack: activations -> bf16; weights -> bf16 transposed [N][K]
  cvt_bf16_kernel<<<(TOKS * HID + 255) / 256, 256, 0, stream>>>(hidden, hb, TOKS * HID);
  cvt_bf16_kernel<<<(TOKS * HID + 255) / 256, 256, 0, stream>>>(mu, mb, TOKS * HID);
  transpose_cvt_kernel<<<dim3(2048 / 32, 2048 / 32), 256, 0, stream>>>(Wq,  WqT,  2048, 2048);
  transpose_cvt_kernel<<<dim3(2048 / 32, 2048 / 32), 256, 0, stream>>>(Wmq, WmqT, 2048, 2048);
  transpose_cvt_kernel<<<dim3(512 / 32, 2048 / 32), 256, 0, stream>>>(Wk,  WkT,  2048, 512);
  transpose_cvt_kernel<<<dim3(512 / 32, 2048 / 32), 256, 0, stream>>>(Wmk, WmkT, 2048, 512);
  transpose_cvt_kernel<<<dim3(512 / 32, 2048 / 32), 256, 0, stream>>>(Wv,  WvT,  2048, 512);
  transpose_cvt_kernel<<<dim3(512 / 32, 2048 / 32), 256, 0, stream>>>(Wmv, WmvT, 2048, 512);
  transpose_cvt_kernel<<<dim3(2048 / 32, 2048 / 32), 256, 0, stream>>>(Wo,  WoT,  2048, 2048);

  // 1) fused projections: X = hidden@W + mu@Wm  (q, k, v)
  gemm_bf16_kernel<<<dim3(TOKS / 64, 2048 / 64), 256, 0, stream>>>(hb, mb, WqT, WmqT, qf, 2048);
  gemm_bf16_kernel<<<dim3(TOKS / 64, 512 / 64), 256, 0, stream>>>(hb, mb, WkT, WmkT, kf, 512);
  gemm_bf16_kernel<<<dim3(TOKS / 64, 512 / 64), 256, 0, stream>>>(hb, mb, WvT, WmvT, vf, 512);

  // 2) RMSNorm + RoPE (q, k); V -> bf16 transposed [NKVH*HD][TOKS]
  norm_rope_kernel<<<TOKS * NHQ, 32, 0, stream>>>(qf, qb, qnw, pos, NHQ);
  norm_rope_kernel<<<TOKS * NKVH, 32, 0, stream>>>(kf, kb, knw, pos, NKVH);
  transpose_cvt_kernel<<<dim3(512 / 32, 2048 / 32), 256, 0, stream>>>(vf, vbT, 2048, 512);

  // 3) causal flash attention with GQA
  flash_attn_kernel<<<dim3(TOKS / 64, NHQ), 128, 0, stream>>>(qb, kb, vbT, ob);

  // 4) output projection
  gemm_bf16_kernel<<<dim3(TOKS / 64, HID / 64), 256, 0, stream>>>(ob, nullptr, WoT, nullptr, out, 2048);
}
